// LongShortAttention_919123001878
// MI455X (gfx1250) — matrix operand order repaired and checked by reading, not compiled, not run on hardware
//
#include <hip/hip_runtime.h>
#include <hip/hip_bf16.h>
#include <math.h>

// ---------------------------------------------------------------------------
// Types / helpers for CDNA5 WMMA (wave32)
// ---------------------------------------------------------------------------
typedef __attribute__((ext_vector_type(16))) __bf16 bf16x16;
typedef __attribute__((ext_vector_type(8)))  float  f32x8;
typedef __attribute__((ext_vector_type(4)))  int    v4i;

union FragU { unsigned int u[8]; bf16x16 v; };
union Pack4 { unsigned short us[4]; unsigned long long ull; };

#ifndef __has_builtin
#define __has_builtin(x) 0
#endif
#if __has_builtin(__builtin_amdgcn_global_load_async_to_lds_b128) && \
    __has_builtin(__builtin_amdgcn_s_wait_asynccnt)
#define USE_ASYNC_LDS 1
#else
#define USE_ASYNC_LDS 0
#endif
#define GLOBAL_AS __attribute__((address_space(1)))
#define LDS_AS    __attribute__((address_space(3)))

#if USE_ASYNC_LDS
// prototype (from hipcc diagnostic): (int4 AS1*, int4 AS3*, imm i32, imm i32)
#define ASYNC_CP_B128(gp, lp) \
  __builtin_amdgcn_global_load_async_to_lds_b128( \
      (GLOBAL_AS v4i*)(gp), (LDS_AS v4i*)(lp), 0, 0)
#endif

__device__ __forceinline__ unsigned short f2bf(float f) {
  unsigned int u = __float_as_uint(f);
  u += 0x7fffu + ((u >> 16) & 1u);          // round-to-nearest-even
  return (unsigned short)(u >> 16);
}

// A-matrix 16x32 bf16 fragment from row-major LDS (row stride in ushorts).
// ISA 7.12.2: lane l -> M = l%16; lanes0-15: V0-3 K=0..7, V4-7 K=16..23;
// lanes16-31: V0-3 K=8..15, V4-7 K=24..31.
__device__ __forceinline__ bf16x16 load_a_frag(const unsigned short* S, int stride,
                                               int mbase, int kbase, int lane) {
  const int m = mbase + (lane & 15);
  const int half = lane >> 4;
  FragU f;
#pragma unroll
  for (int g = 0; g < 8; ++g) {
    const int k0 = kbase + ((g < 4) ? 0 : 16) + half * 8 + (g & 3) * 2;
    f.u[g] = *(const unsigned int*)(S + m * stride + k0);
  }
  return f.v;
}

// B-matrix 32x16 bf16 fragment from K-contiguous (transposed) LDS T[n][k].
// Same K order as A: lanes0-15: V0-3 K=0..7, V4-7 K=16..23; lanes16-31: V0-3 K=8..15, V4-7 K=24..31.
__device__ __forceinline__ bf16x16 load_b_frag(const unsigned short* T, int stride,
                                               int nbase, int kbase, int lane) {
  const int n = nbase + (lane & 15);
  const int half = lane >> 4;
  FragU f;
#pragma unroll
  for (int g = 0; g < 8; ++g) {
    const int k0 = kbase + ((g < 4) ? 0 : 16) + half * 8 + (g & 3) * 2;
    f.u[g] = *(const unsigned int*)(T + n * stride + k0);
  }
  return f.v;
}

__device__ __forceinline__ f32x8 wmma_bf16(bf16x16 a, bf16x16 b, f32x8 c) {
  // (neg_a, A, neg_b, B, c_mod, C, reuse_a, reuse_b)
  return __builtin_amdgcn_wmma_f32_16x16x32_bf16(false, a, false, b, (short)0, c,
                                                 false, false);
}

// Problem constants
#define NTOK   4096
#define DIM    1024
#define HEADS  16
#define DH     64
#define WIN    128
#define NSEG   256            // per sequence (seg size 16, r=1 -> R=256)
#define BH     32             // b * heads
#define MTOT   8192           // b * n
#define LN10K  9.2103403719761836f

// ---------------------------------------------------------------------------
// K1: QKV projection (x @ w) + RoPE.  q written bf16 (feeds bf16 WMMA only),
// kv written fp32 (feeds LN / segment-softmax in fp32).
// grid (64, 16, 2)  block 256
// ---------------------------------------------------------------------------
__global__ __launch_bounds__(256) void qkv_rope_kernel(
    const float* __restrict__ x, const float* __restrict__ w_q,
    const float* __restrict__ w_kv, unsigned short* __restrict__ qout,
    float* __restrict__ kvout) {
  __shared__ unsigned short As[128][72];   // bf16 x tile, row-major (8B-aligned rows)
  __shared__ unsigned short Bs[64][72];    // bf16 w tile, transposed [n][k]
  __shared__ float          Cs[128][65];   // fp32 epilogue tile

  const int tid = threadIdx.x;
  const int lane = tid & 31, wave = tid >> 5;
  const int m0 = blockIdx.x * 128;
  const int head = blockIdx.y;
  const int sel = blockIdx.z;
  const float* w = sel ? w_kv : w_q;

  const int wm = (wave & 3) * 32;          // 4 M-groups of 32
  const int wn = (wave >> 2) * 32;         // 2 N-groups of 32
  f32x8 acc[2][2] = {};

  for (int kt = 0; kt < DIM; kt += 64) {
    if (kt + 64 < DIM) {                   // prefetch next K-tile
      __builtin_prefetch(x + (size_t)(m0 + (tid >> 1)) * DIM + kt + 64 + (tid & 1) * 32, 0, 1);
      if (tid < 64) __builtin_prefetch(w + (size_t)(kt + 64 + tid) * DIM + head * DH, 0, 1);
    }
    for (int i = tid; i < (128 * 64) / 4; i += 256) {     // float4 loads, b64 stores
      const int r = i >> 4, c = (i & 15) * 4;
      const float4 v = *(const float4*)(x + (size_t)(m0 + r) * DIM + kt + c);
      Pack4 p;
      p.us[0] = f2bf(v.x); p.us[1] = f2bf(v.y); p.us[2] = f2bf(v.z); p.us[3] = f2bf(v.w);
      *(unsigned long long*)&As[r][c] = p.ull;
    }
    for (int i = tid; i < (64 * 64) / 4; i += 256) {      // transpose W into [n][k]
      const int k = i >> 4, n = (i & 15) * 4;
      const float4 v = *(const float4*)(w + (size_t)(kt + k) * DIM + head * DH + n);
      Bs[n][k] = f2bf(v.x); Bs[n + 1][k] = f2bf(v.y);
      Bs[n + 2][k] = f2bf(v.z); Bs[n + 3][k] = f2bf(v.w);
    }
    __syncthreads();
#pragma unroll
    for (int kk = 0; kk < 64; kk += 32) {
      const bf16x16 a0 = load_a_frag(&As[0][0], 72, wm,      kk, lane);
      const bf16x16 a1 = load_a_frag(&As[0][0], 72, wm + 16, kk, lane);
      const bf16x16 b0 = load_b_frag(&Bs[0][0], 72, wn,      kk, lane);
      const bf16x16 b1 = load_b_frag(&Bs[0][0], 72, wn + 16, kk, lane);
      acc[0][0] = wmma_bf16(a0, b0, acc[0][0]);
      acc[0][1] = wmma_bf16(a0, b1, acc[0][1]);
      acc[1][0] = wmma_bf16(a1, b0, acc[1][0]);
      acc[1][1] = wmma_bf16(a1, b1, acc[1][1]);
    }
    __syncthreads();
  }

  const int half = lane >> 4, ln15 = lane & 15;
#pragma unroll
  for (int i = 0; i < 2; ++i)
#pragma unroll
    for (int j = 0; j < 2; ++j)
#pragma unroll
      for (int g = 0; g < 8; ++g)
        Cs[wm + i * 16 + g + half * 8][wn + j * 16 + ln15] = acc[i][j][g];
  __syncthreads();

  // RoPE: out[d] = v[d]*cos + (d<32 ? -v[d+32] : v[d-32])*sin ; q also * dh^-0.5
  for (int i = tid; i < 128 * 64; i += 256) {
    const int r = i >> 6, d = i & 63;
    const int gm = m0 + r;
    const int t = gm & (NTOK - 1);
    const int bi = gm >> 12;
    const int j = d & 31;
    const float freq = __expf(-(float)j * (LN10K / 32.0f));
    float sn, cs;
    __sincosf((float)t * freq, &sn, &cs);
    const float v0 = Cs[r][d];
    const float vo = (d < 32) ? -Cs[r][d + 32] : Cs[r][d - 32];
    const float rv = v0 * cs + vo * sn;
    const size_t idx = (((size_t)(bi * HEADS + head)) * NTOK + t) * DH + d;
    if (sel == 0) qout[idx] = f2bf(rv * 0.125f);
    else          kvout[idx] = rv;
  }
}

// ---------------------------------------------------------------------------
// K2: per-segment low-rank projection (softmax over s=16) + LayerNorm -> bf16.
// grid 8192 (bh*nseg), block 64 (one thread per dh column)
// ---------------------------------------------------------------------------
__global__ __launch_bounds__(64) void global_kv_kernel(
    const float* __restrict__ kv, const float* __restrict__ w_proj,
    const float* __restrict__ gg, const float* __restrict__ gb,
    unsigned short* __restrict__ gkv) {
  const int blk = blockIdx.x;
  const int bh = blk >> 8, seg = blk & 255;
  const int d = threadIdx.x;               // 0..63

  __shared__ float red[16][64];
  __shared__ float pr[16];
  __shared__ float t0[64];
  __shared__ float mv[2];

  const float wp = w_proj[d];
  float v[16];
  const float* base = kv + ((size_t)bh * NTOK + seg * 16) * DH;
#pragma unroll
  for (int s = 0; s < 16; ++s) v[s] = base[s * DH + d];
#pragma unroll
  for (int s = 0; s < 16; ++s) red[s][d] = v[s] * wp;
  __syncthreads();
  if (d < 16) {
    float a = 0.f;
#pragma unroll
    for (int j = 0; j < 64; ++j) a += red[d][j];
    red[d][0] = a;                         // logit s = d
  }
  __syncthreads();
  if (d == 0) {
    float mx = -1e30f;
    for (int s = 0; s < 16; ++s) mx = fmaxf(mx, red[s][0]);
    float ssum = 0.f;
    for (int s = 0; s < 16; ++s) { float e = __expf(red[s][0] - mx); pr[s] = e; ssum += e; }
    const float inv = 1.0f / ssum;
    for (int s = 0; s < 16; ++s) pr[s] *= inv;
  }
  __syncthreads();
  float o = 0.f;
#pragma unroll
  for (int s = 0; s < 16; ++s) o += pr[s] * v[s];
  t0[d] = o;
  __syncthreads();
  if (d == 0) {
    float s1 = 0.f, s2 = 0.f;
    for (int j = 0; j < 64; ++j) { s1 += t0[j]; s2 += t0[j] * t0[j]; }
    const float mean = s1 * (1.0f / 64.0f);
    mv[0] = mean;
    mv[1] = s2 * (1.0f / 64.0f) - mean * mean;
  }
  __syncthreads();
  const float r = (o - mv[0]) * rsqrtf(mv[1] + 1e-5f) * gg[d] + gb[d];
  gkv[((size_t)bh * NSEG + seg) * DH + d] = f2bf(r);
}

// ---------------------------------------------------------------------------
// K3: fused long-short attention with online (flash) softmax.
// Keys == values: 8 chunks of 64 keys = [256 global | prev window | cur window].
// grid (32 windows, 32 bh), block 256 (8 waves; wave owns 16 query rows)
// ---------------------------------------------------------------------------
__global__ __launch_bounds__(256) void attn_kernel(
    const unsigned short* __restrict__ qb, const float* __restrict__ kv,
    const unsigned short* __restrict__ gkvb, const float* __restrict__ lng,
    const float* __restrict__ lnb, unsigned short* __restrict__ ctx) {
  __shared__ unsigned short Qs[128][64];       // q tile bf16 (A source)
  __shared__ unsigned short Kd[64][64];        // key chunk [key][dh]   (B for QK^T)
  __shared__ unsigned short Vt[64][64];        // key chunk [dh][key]   (B for PV)
  __shared__ unsigned short Ps[8][16][64];     // per-wave P bf16 (A source)

  const int tid = threadIdx.x, lane = tid & 31, wave = tid >> 5;
  const int win = blockIdx.x, bh = blockIdx.y;
  const int half = lane >> 4, ln15 = lane & 15;
  const int qr = wave * 16;

  // ---- stage Q tile (bf16 in HBM -> direct/async copy)
#if USE_ASYNC_LDS
  for (int i = tid; i < 128 * 8; i += 256) {
    const int r = i >> 3, c8 = (i & 7) * 8;
    ASYNC_CP_B128(qb + ((size_t)bh * NTOK + win * WIN + r) * DH + c8, &Qs[r][c8]);
  }
  __builtin_amdgcn_s_wait_asynccnt(0);
#else
  for (int i = tid; i < 128 * 8; i += 256) {
    const int r = i >> 3, c8 = (i & 7) * 8;
    *(unsigned long long*)&Qs[r][c8] =
        *(const unsigned long long*)(qb + ((size_t)bh * NTOK + win * WIN + r) * DH + c8);
  }
#endif
  __syncthreads();

  f32x8 outacc[4] = {};
  float rowmax[8], rowsum[8];
#pragma unroll
  for (int g = 0; g < 8; ++g) { rowmax[g] = -1e30f; rowsum[g] = 0.f; }

  for (int c = 0; c < 8; ++c) {
    if (c >= 4 && c < 6 && win == 0) continue;   // window 0: prev = padding (fully masked)

    // ---- stage 64 keys into Kd (row-major) and Vt (transposed)
    if (c < 4) {                                  // global keys (bf16 in HBM)
#if USE_ASYNC_LDS
      for (int i = tid; i < 64 * 8; i += 256) {
        const int j = i >> 3, c8 = (i & 7) * 8;
        ASYNC_CP_B128(gkvb + ((size_t)bh * NSEG + c * 64 + j) * DH + c8, &Kd[j][c8]);
      }
      __builtin_amdgcn_s_wait_asynccnt(0);
#else
      for (int i = tid; i < 64 * 8; i += 256) {
        const int j = i >> 3, c8 = (i & 7) * 8;
        *(unsigned long long*)&Kd[j][c8] =
            *(const unsigned long long*)(gkvb + ((size_t)bh * NSEG + c * 64 + j) * DH + c8);
      }
#endif
      __syncthreads();
      for (int i = tid; i < 64 * 64; i += 256) {  // LDS->LDS transpose
        const int j = i >> 6, d = i & 63;
        Vt[d][j] = Kd[j][d];
      }
    } else {                                      // local keys: LayerNorm on the fly
      if (tid < 64) {
        const int j = tid;
        const int tt = (win - 1) * WIN + (c - 4) * 64 + j;
        const float4* rp = (const float4*)(kv + ((size_t)bh * NTOK + tt) * DH);
        float4 vv[16];
        float s1 = 0.f, s2 = 0.f;
#pragma unroll
        for (int d4 = 0; d4 < 16; ++d4) {
          vv[d4] = rp[d4];
          s1 += vv[d4].x + vv[d4].y + vv[d4].z + vv[d4].w;
          s2 += vv[d4].x * vv[d4].x + vv[d4].y * vv[d4].y +
                vv[d4].z * vv[d4].z + vv[d4].w * vv[d4].w;
        }
        const float mean = s1 * (1.0f / 64.0f);
        const float var = s2 * (1.0f / 64.0f) - mean * mean;
        const float rs = rsqrtf(var + 1e-5f);
#pragma unroll
        for (int d4 = 0; d4 < 16; ++d4) {
          const float e[4] = {vv[d4].x, vv[d4].y, vv[d4].z, vv[d4].w};
          Pack4 p;
#pragma unroll
          for (int q4 = 0; q4 < 4; ++q4) {
            const int d = d4 * 4 + q4;
            const float nv = (e[q4] - mean) * rs * lng[d] + lnb[d];
            const unsigned short bv = f2bf(nv);
            p.us[q4] = bv;
            Vt[d][j] = bv;
          }
          *(unsigned long long*)&Kd[j][d4 * 4] = p.ull;
        }
      }
    }
    __syncthreads();

    // ---- sim = Q(16x64) @ K^T(64x64-chunk)
    const bf16x16 a0 = load_a_frag(&Qs[0][0], 64, qr, 0, lane);
    const bf16x16 a1 = load_a_frag(&Qs[0][0], 64, qr, 32, lane);
    f32x8 s[4];
#pragma unroll
    for (int j2 = 0; j2 < 4; ++j2) {
      f32x8 z = {};
      const bf16x16 b0 = load_b_frag(&Kd[0][0], 64, j2 * 16, 0, lane);
      const bf16x16 b1 = load_b_frag(&Kd[0][0], 64, j2 * 16, 32, lane);
      z = wmma_bf16(a0, b0, z);
      s[j2] = wmma_bf16(a1, b1, z);
    }

    // ---- masks (C-frag layout: row = g + half*8, col = j2*16 + ln15)
    const float MASKV = -1e30f;
#pragma unroll
    for (int j2 = 0; j2 < 4; ++j2)
#pragma unroll
      for (int g = 0; g < 8; ++g) {
        const int mrow = g + half * 8;
        const int qpos = win * WIN + qr + mrow;
        bool ok;
        if (c < 4) {                       // global causal: qpos >= seg_max
          const int gk = c * 64 + j2 * 16 + ln15;
          ok = qpos >= gk * 16 + 15;
        } else if (c < 6) {                // prev window: fully visible
          ok = true;
        } else {                           // cur window causal: key_j <= query_i
          const int jl = (c - 6) * 64 + j2 * 16 + ln15;
          ok = jl <= (qr + mrow);
        }
        if (!ok) s[j2][g] = MASKV;
      }

    // ---- online softmax (rows live in 16-lane halves -> xor-shuffles 1/2/4/8)
    float mloc[8];
#pragma unroll
    for (int g = 0; g < 8; ++g) mloc[g] = MASKV;
#pragma unroll
    for (int j2 = 0; j2 < 4; ++j2)
#pragma unroll
      for (int g = 0; g < 8; ++g) mloc[g] = fmaxf(mloc[g], s[j2][g]);
#pragma unroll
    for (int off = 1; off < 16; off <<= 1)
#pragma unroll
      for (int g = 0; g < 8; ++g)
        mloc[g] = fmaxf(mloc[g], __shfl_xor(mloc[g], off, 32));

    float corr[8], psum[8];
#pragma unroll
    for (int g = 0; g < 8; ++g) {
      const float nm = fmaxf(rowmax[g], mloc[g]);
      corr[g] = __expf(rowmax[g] - nm);
      rowmax[g] = nm;
      psum[g] = 0.f;
    }
#pragma unroll
    for (int j2 = 0; j2 < 4; ++j2)
#pragma unroll
      for (int g = 0; g < 8; ++g) {
        const float p = __expf(s[j2][g] - rowmax[g]);
        s[j2][g] = p;
        psum[g] += p;
      }
#pragma unroll
    for (int off = 1; off < 16; off <<= 1)
#pragma unroll
      for (int g = 0; g < 8; ++g) psum[g] += __shfl_xor(psum[g], off, 32);
#pragma unroll
    for (int g = 0; g < 8; ++g) rowsum[g] = rowsum[g] * corr[g] + psum[g];
#pragma unroll
    for (int t = 0; t < 4; ++t)
#pragma unroll
      for (int g = 0; g < 8; ++g) outacc[t][g] *= corr[g];

    // ---- P (C-layout) -> wave-private LDS -> A-layout
#pragma unroll
    for (int j2 = 0; j2 < 4; ++j2)
#pragma unroll
      for (int g = 0; g < 8; ++g)
        Ps[wave][g + half * 8][j2 * 16 + ln15] = f2bf(s[j2][g]);
    __syncthreads();

    // ---- out += P(16x64) @ V(64x64chunk -> dh)
    const bf16x16 pa0 = load_a_frag(&Ps[wave][0][0], 64, 0, 0, lane);
    const bf16x16 pa1 = load_a_frag(&Ps[wave][0][0], 64, 0, 32, lane);
#pragma unroll
    for (int t = 0; t < 4; ++t) {
      const bf16x16 vb0 = load_b_frag(&Vt[0][0], 64, t * 16, 0, lane);
      const bf16x16 vb1 = load_b_frag(&Vt[0][0], 64, t * 16, 32, lane);
      outacc[t] = wmma_bf16(pa0, vb0, outacc[t]);
      outacc[t] = wmma_bf16(pa1, vb1, outacc[t]);
    }
    __syncthreads();
  }

  // ---- epilogue: normalize, write bf16 ctx in [b, n, h*dh] layout
  const int bi = bh >> 4, head = bh & 15;
#pragma unroll
  for (int t = 0; t < 4; ++t)
#pragma unroll
    for (int g = 0; g < 8; ++g) {
      const int mrow = g + half * 8;
      const int tok = win * WIN + qr + mrow;
      const float v = outacc[t][g] / rowsum[g];
      ctx[((size_t)(bi * NTOK + tok)) * DIM + head * DH + t * 16 + ln15] = f2bf(v);
    }
}

// ---------------------------------------------------------------------------
// K4: output projection ctx(bf16) @ w_out + b_out -> fp32.
// grid (64, 16) block 256
// ---------------------------------------------------------------------------
__global__ __launch_bounds__(256) void out_proj_kernel(
    const unsigned short* __restrict__ ctx, const float* __restrict__ w_out,
    const float* __restrict__ b_out, float* __restrict__ out) {
  __shared__ unsigned short As[128][72];
  __shared__ unsigned short Bs[64][72];

  const int tid = threadIdx.x;
  const int lane = tid & 31, wave = tid >> 5;
  const int m0 = blockIdx.x * 128;
  const int n0 = blockIdx.y * 64;

  const int wm = (wave & 3) * 32;
  const int wn = (wave >> 2) * 32;
  f32x8 acc[2][2] = {};

  for (int kt = 0; kt < DIM; kt += 64) {
    if (kt + 64 < DIM && tid < 64)
      __builtin_prefetch(w_out + (size_t)(kt + 64 + tid) * DIM + n0, 0, 1);
    // A tile: ctx already bf16 -> direct/async copy (16B chunks, aligned: row=144B)
#if USE_ASYNC_LDS
    for (int i = tid; i < 128 * 8; i += 256) {
      const int r = i >> 3, c8 = (i & 7) * 8;
      ASYNC_CP_B128(ctx + (size_t)(m0 + r) * DIM + kt + c8, &As[r][c8]);
    }
#else
    for (int i = tid; i < 128 * 8; i += 256) {
      const int r = i >> 3, c8 = (i & 7) * 8;
      *(unsigned long long*)&As[r][c8] =
          *(const unsigned long long*)(ctx + (size_t)(m0 + r) * DIM + kt + c8);
    }
#endif
    for (int i = tid; i < (64 * 64) / 4; i += 256) {
      const int k = i >> 4, n = (i & 15) * 4;
      const float4 v = *(const float4*)(w_out + (size_t)(kt + k) * DIM + n0 + n);
      Bs[n][k] = f2bf(v.x); Bs[n + 1][k] = f2bf(v.y);
      Bs[n + 2][k] = f2bf(v.z); Bs[n + 3][k] = f2bf(v.w);
    }
#if USE_ASYNC_LDS
    __builtin_amdgcn_s_wait_asynccnt(0);
#endif
    __syncthreads();
#pragma unroll
    for (int kk = 0; kk < 64; kk += 32) {
      const bf16x16 a0 = load_a_frag(&As[0][0], 72, wm,      kk, lane);
      const bf16x16 a1 = load_a_frag(&As[0][0], 72, wm + 16, kk, lane);
      const bf16x16 b0 = load_b_frag(&Bs[0][0], 72, wn,      kk, lane);
      const bf16x16 b1 = load_b_frag(&Bs[0][0], 72, wn + 16, kk, lane);
      acc[0][0] = wmma_bf16(a0, b0, acc[0][0]);
      acc[0][1] = wmma_bf16(a0, b1, acc[0][1]);
      acc[1][0] = wmma_bf16(a1, b0, acc[1][0]);
      acc[1][1] = wmma_bf16(a1, b1, acc[1][1]);
    }
    __syncthreads();
  }

  const int half = lane >> 4, ln15 = lane & 15;
#pragma unroll
  for (int i = 0; i < 2; ++i)
#pragma unroll
    for (int j = 0; j < 2; ++j) {
      const int col = wn + j * 16 + ln15;
      const float bias = b_out[n0 + col];
#pragma unroll
      for (int g = 0; g < 8; ++g) {
        const int row = wm + i * 16 + g + half * 8;
        out[(size_t)(m0 + row) * DIM + n0 + col] = acc[i][j][g] + bias;
      }
    }
}

// ---------------------------------------------------------------------------
// Host launcher
// ---------------------------------------------------------------------------
extern "C" void kernel_launch(void* const* d_in, const int* in_sizes, int n_in,
                              void* d_out, int out_size, void* d_ws, size_t ws_size,
                              hipStream_t stream) {
  (void)in_sizes; (void)n_in; (void)out_size; (void)ws_size;
  const float* x           = (const float*)d_in[0];
  const float* w_q         = (const float*)d_in[1];
  const float* w_kv        = (const float*)d_in[2];
  const float* w_proj      = (const float*)d_in[3];
  const float* ln_local_g  = (const float*)d_in[4];
  const float* ln_local_b  = (const float*)d_in[5];
  const float* ln_global_g = (const float*)d_in[6];
  const float* ln_global_b = (const float*)d_in[7];
  const float* w_out       = (const float*)d_in[8];
  const float* b_out       = (const float*)d_in[9];
  float* outp = (float*)d_out;

  // Workspace layout: q bf16 (16MB), kv fp32 (32MB), gkv bf16 (1MB), ctx bf16 (16MB)
  unsigned short* qb   = (unsigned short*)d_ws;
  float*          kvf  = (float*)(qb + (size_t)BH * NTOK * DH);
  unsigned short* gkvb = (unsigned short*)(kvf + (size_t)BH * NTOK * DH);
  unsigned short* ctx  = gkvb + (size_t)BH * NSEG * DH;

  qkv_rope_kernel<<<dim3(MTOT / 128, HEADS, 2), 256, 0, stream>>>(x, w_q, w_kv, qb, kvf);
  global_kv_kernel<<<dim3(BH * NSEG), 64, 0, stream>>>(kvf, w_proj, ln_global_g,
                                                       ln_global_b, gkvb);
  attn_kernel<<<dim3(NTOK / WIN, BH), 256, 0, stream>>>(qb, kvf, gkvb, ln_local_g,
                                                        ln_local_b, ctx);
  out_proj_kernel<<<dim3(MTOT / 128, DIM / 64), 256, 0, stream>>>(ctx, w_out,
                                                                  b_out, outp);
}
